// GPT_22445499089702
// MI455X (gfx1250) — compile-verified, matrix operand order checked by reading, not run on hardware
//
#include <hip/hip_runtime.h>
#include <hip/hip_bf16.h>

typedef __bf16 bf16_t;
typedef __attribute__((ext_vector_type(16))) __bf16 v16bf;
typedef __attribute__((ext_vector_type(8)))  __bf16 v8bf;
typedef __attribute__((ext_vector_type(8)))  float  v8f;
typedef __attribute__((address_space(3))) bf16_t lds_bf16_t;

// ---------------------------------------------------------------------------
// CDNA5 primitives
// ---------------------------------------------------------------------------
__device__ inline v8f wmma_bf16(v16bf a, v16bf b, v8f c) {
  return __builtin_amdgcn_wmma_f32_16x16x32_bf16(
      /*neg_a=*/false, a, /*neg_b=*/false, b,
      /*c_mod=*/(short)0, c, /*reuse_a=*/false, /*reuse_b=*/false);
}

// byte offset of a __shared__ object within the workgroup LDS aperture
__device__ inline uint32_t lds_addr_of(const bf16_t* p) {
  return (uint32_t)(uint64_t)(lds_bf16_t*)p;
}

// async DMA: 16 bytes global -> LDS per lane, tracked by ASYNCcnt
__device__ inline void async_b128(uint32_t lds_byte, const bf16_t* g) {
  asm volatile("global_load_async_to_lds_b128 %0, %1, off"
               :: "v"(lds_byte), "v"(g) : "memory");
}
__device__ inline void wait_async0() {
  asm volatile("s_wait_asynccnt 0x0" ::: "memory");
}

// Load a 16(row) x 32(K) bf16 fragment (A layout; also B^T for the B operand).
// ISA wave32 layout: lanes 0-15 -> row=lane, Kbase=0 ; lanes 16-31 -> row-16, Kbase=8.
// e[0..7] = K kb..kb+7 ; e[8..15] = K kb+16..kb+23.
__device__ inline v16bf load_frag(const bf16_t* __restrict__ src, int ld) {
  int lane = threadIdx.x & 31;
  int row  = lane & 15;
  int kb   = (lane >> 4) << 3;
  const bf16_t* p = src + (size_t)row * ld + kb;
  union { v16bf v; v8bf h[2]; } u;
  u.h[0] = *(const v8bf*)(p);
  u.h[1] = *(const v8bf*)(p + 16);
  return u.v;
}

__device__ inline float gelu_f(float x) {
  const float c = 0.7978845608028654f;  // sqrt(2/pi)
  float x3 = x * x * x;
  return 0.5f * x * (1.0f + tanhf(c * (x + 0.044715f * x3)));
}

// ---------------------------------------------------------------------------
// bf16 WMMA GEMM:  C[M,N] = epilogue(A[M,K] @ Bt^T + bias, res)
//   A  : [M,K] bf16 row-major
//   Bt : [N,K] bf16 row-major (all weights pre-transposed at convert time)
// EPI: 0 = store f32 (no bias)            (LM head)
//      1 = +bias, store bf16              (QKV)
//      2 = +bias +res(f32), store f32     (out-proj / FC2 residual)
//      3 = +bias, GELU, store bf16        (FC1)
// Block tile 128x128x32, 8 waves (2m x 4n), wave tile 64x32 = 4x2 WMMA tiles.
// Tiles staged with double-buffered async global->LDS DMA.
// ---------------------------------------------------------------------------
#define BM 128
#define BN 128
#define BKK 32
#define LDT 40  // padded LDS row stride (bf16), multiple of 8

template <int EPI>
__global__ __launch_bounds__(256) void gemm_bf16(
    const bf16_t* __restrict__ A, const bf16_t* __restrict__ Bt,
    const float* __restrict__ bias, const float* __restrict__ res,
    void* __restrict__ Cout, int M, int N, int K) {
  __shared__ bf16_t As[2][BM * LDT];
  __shared__ bf16_t Bs[2][BN * LDT];

  const int tid  = threadIdx.x;
  const int wave = tid >> 5;
  const int wm   = (wave >> 2) * 64;  // 0 or 64
  const int wn   = (wave & 3) * 32;   // 0,32,64,96
  const int bm0  = blockIdx.y * BM;
  const int bn0  = blockIdx.x * BN;

  // staging role: thread -> (row 0..127, col 0/16) of both 128x32 tiles
  const int lr = tid >> 1;
  const int lc = (tid & 1) * 16;
  uint32_t asb[2], bsb[2];
  asb[0] = lds_addr_of(&As[0][lr * LDT + lc]);
  asb[1] = lds_addr_of(&As[1][lr * LDT + lc]);
  bsb[0] = lds_addr_of(&Bs[0][lr * LDT + lc]);
  bsb[1] = lds_addr_of(&Bs[1][lr * LDT + lc]);
  const bf16_t* ga = A + (size_t)(bm0 + lr) * K + lc;
  const int gbn = bn0 + lr;
  const bf16_t* gb = Bt + (size_t)gbn * K + lc;
  const bool bn_ok = gbn < N;  // LM-head edge tiles

  auto stage = [&](int k0, int buf) {
    async_b128(asb[buf],      ga + k0);
    async_b128(asb[buf] + 16, ga + k0 + 8);
    if (bn_ok) {
      async_b128(bsb[buf],      gb + k0);
      async_b128(bsb[buf] + 16, gb + k0 + 8);
    } else {
      v8bf z;
#pragma unroll
      for (int i = 0; i < 8; ++i) z[i] = (bf16_t)0.0f;
      *(v8bf*)&Bs[buf][lr * LDT + lc]     = z;
      *(v8bf*)&Bs[buf][lr * LDT + lc + 8] = z;
    }
  };

  v8f acc[4][2] = {};
  stage(0, 0);
  int cur = 0;
  for (int k0 = 0; k0 < K; k0 += BKK) {
    wait_async0();       // my async DMAs for buf `cur` landed
    __syncthreads();     // everyone's landed; everyone done reading buf cur^1
    if (k0 + BKK < K) stage(k0 + BKK, cur ^ 1);

    v16bf bfr[2];
#pragma unroll
    for (int ni = 0; ni < 2; ++ni)
      bfr[ni] = load_frag(&Bs[cur][(wn + ni * 16) * LDT], LDT);
#pragma unroll
    for (int mi = 0; mi < 4; ++mi) {
      v16bf af = load_frag(&As[cur][(wm + mi * 16) * LDT], LDT);
#pragma unroll
      for (int ni = 0; ni < 2; ++ni)
        acc[mi][ni] = wmma_bf16(af, bfr[ni], acc[mi][ni]);
    }
    cur ^= 1;
  }

  // --- epilogue ---
  const int lane = tid & 31;
  const int half = lane >> 4;
  const int nl   = lane & 15;
#pragma unroll
  for (int mi = 0; mi < 4; ++mi)
#pragma unroll
    for (int ni = 0; ni < 2; ++ni)
#pragma unroll
      for (int r = 0; r < 8; ++r) {
        int m = bm0 + wm + mi * 16 + r + half * 8;
        int n = bn0 + wn + ni * 16 + nl;
        if (n < N) {
          float v = acc[mi][ni][r];
          if (EPI != 0) v += bias[n];
          if (EPI == 3) v = gelu_f(v);
          if (EPI == 2) v += res[(size_t)m * N + n];
          if (EPI == 1 || EPI == 3)
            ((bf16_t*)Cout)[(size_t)m * N + n] = (bf16_t)v;
          else
            ((float*)Cout)[(size_t)m * N + n] = v;
        }
      }
}

// ---------------------------------------------------------------------------
// Embedding: x[b,s,:] = tok[id] + pos[s]   (f32)
// ---------------------------------------------------------------------------
__global__ void embed_kernel(const int* __restrict__ ids,
                             const float* __restrict__ tok,
                             const float* __restrict__ pos,
                             float* __restrict__ x) {
  int bs = blockIdx.x;
  int s  = bs & 1023;
  int id = ids[bs];
  for (int i = threadIdx.x; i < 768; i += blockDim.x)
    x[(size_t)bs * 768 + i] = tok[(size_t)id * 768 + i] + pos[(size_t)s * 768 + i];
}

// ---------------------------------------------------------------------------
// fp32 -> bf16 bulk convert (tok_embed, already [N,K])
// ---------------------------------------------------------------------------
__global__ void f32_to_bf16(const float* __restrict__ in,
                            bf16_t* __restrict__ out, size_t n) {
  size_t i = (size_t)blockIdx.x * blockDim.x + threadIdx.x;
  if (i < n) out[i] = (bf16_t)in[i];
}

// ---------------------------------------------------------------------------
// fp32 [K,N] -> bf16 [N,K] transposed convert (LDS-tiled, K,N multiples of 32)
// ---------------------------------------------------------------------------
__global__ __launch_bounds__(256) void f32_to_bf16_t(
    const float* __restrict__ in, bf16_t* __restrict__ out, int K, int N) {
  __shared__ float t[32][33];
  int kb = blockIdx.y * 32, nb = blockIdx.x * 32;
  int tx = threadIdx.x & 31, ty = threadIdx.x >> 5;  // 32 x 8
  for (int i = ty; i < 32; i += 8)
    t[i][tx] = in[(size_t)(kb + i) * N + nb + tx];
  __syncthreads();
  for (int i = ty; i < 32; i += 8)
    out[(size_t)(nb + i) * K + kb + tx] = (bf16_t)t[tx][i];
}

// ---------------------------------------------------------------------------
// LayerNorm (row-per-block), f32 in, bf16 out
// ---------------------------------------------------------------------------
__global__ __launch_bounds__(256) void layernorm_bf16(
    const float* __restrict__ x, const float* __restrict__ g,
    const float* __restrict__ bta, bf16_t* __restrict__ out, int D) {
  int row = blockIdx.x;
  const float* xr = x + (size_t)row * D;
  __shared__ float rs[8], rs2[8];
  int tid = threadIdx.x;
  float s = 0.f, s2 = 0.f;
  for (int i = tid; i < D; i += 256) {
    float v = xr[i];
    s += v; s2 += v * v;
  }
#pragma unroll
  for (int o = 16; o; o >>= 1) {
    s  += __shfl_xor(s, o, 32);
    s2 += __shfl_xor(s2, o, 32);
  }
  if ((tid & 31) == 0) { rs[tid >> 5] = s; rs2[tid >> 5] = s2; }
  __syncthreads();
  s = 0.f; s2 = 0.f;
#pragma unroll
  for (int i = 0; i < 8; ++i) { s += rs[i]; s2 += rs2[i]; }
  float mean = s / D;
  float var  = s2 / D - mean * mean;
  float rstd = rsqrtf(var + 1e-5f);
  for (int i = tid; i < D; i += 256)
    out[(size_t)row * D + i] = (bf16_t)((xr[i] - mean) * rstd * g[i] + bta[i]);
}

// ---------------------------------------------------------------------------
// Reshape qkv [B*S, 2304] bf16 -> Qh,Kh [BH,S,64], Vt [BH,64,S]
// ---------------------------------------------------------------------------
__global__ void reshape_qkv(const bf16_t* __restrict__ qkv,
                            bf16_t* __restrict__ Qh, bf16_t* __restrict__ Kh,
                            bf16_t* __restrict__ Vt) {
  int bs = blockIdx.x;
  int b = bs >> 10, s = bs & 1023;
  const bf16_t* src = qkv + (size_t)bs * 2304;
  for (int i = threadIdx.x; i < 768; i += blockDim.x) {
    int h = i >> 6, d = i & 63;
    size_t bh   = (size_t)(b * 12 + h);
    size_t bhsd = (bh * 1024 + s) * 64 + d;
    Qh[bhsd] = src[i];
    Kh[bhsd] = src[768 + i];
    Vt[(bh * 64 + d) * 1024 + s] = src[1536 + i];
  }
}

// ---------------------------------------------------------------------------
// Fused attention per (b,h, 16-row q-tile):
//   scores = (Q K^T)/8 + causal mask (WMMA) -> LDS f32 -> softmax -> bf16 P
//   O = P V (WMMA, V pre-transposed) -> attn_out bf16 [B*S, 768]
// ---------------------------------------------------------------------------
__global__ __launch_bounds__(256) void attn_kernel(
    const bf16_t* __restrict__ Qh, const bf16_t* __restrict__ Kh,
    const bf16_t* __restrict__ Vt, bf16_t* __restrict__ Out) {
  const int S = 1024, HD = 64, H = 12;
  const int LP = 1032;
  int qt = blockIdx.x;
  int bh = blockIdx.y;
  int b = bh / H, h = bh % H;
  int q0 = qt * 16;

  __shared__ bf16_t qs[16 * 72];
  __shared__ float  sc[16 * 1024];
  __shared__ bf16_t pb[16 * 1032];

  int tid = threadIdx.x;
  {  // stage Q tile (16x64)
    const bf16_t* qsrc = Qh + ((size_t)bh * S + q0) * HD;
    int r = tid >> 4;
    int c = (tid & 15) * 4;
#pragma unroll
    for (int i = 0; i < 4; ++i) qs[r * 72 + c + i] = qsrc[r * HD + c + i];
  }
  __syncthreads();

  int wave = tid >> 5, lane = tid & 31;
  int half = lane >> 4, nl = lane & 15;

  // phase 1: scores, 8 waves x 8 kv-tiles each
  v16bf qf0 = load_frag(qs, 72);
  v16bf qf1 = load_frag(qs + 32, 72);
  for (int nt = wave; nt < 64; nt += 8) {
    const bf16_t* kb = Kh + ((size_t)bh * S + nt * 16) * HD;
    v8f a = {};
    a = wmma_bf16(qf0, load_frag(kb, HD), a);
    a = wmma_bf16(qf1, load_frag(kb + 32, HD), a);
#pragma unroll
    for (int r = 0; r < 8; ++r) {
      int m  = r + half * 8;
      int kv = nt * 16 + nl;
      float v = a[r] * 0.125f;
      if (kv > q0 + m) v = -1e30f;
      sc[m * 1024 + kv] = v;
    }
  }
  __syncthreads();

  // phase 2: row softmax, 16 threads per row
  {
    int r = tid >> 4, lid = tid & 15;
    float mx = -1e30f;
    for (int c = lid; c < 1024; c += 16) mx = fmaxf(mx, sc[r * 1024 + c]);
#pragma unroll
    for (int o = 8; o; o >>= 1) mx = fmaxf(mx, __shfl_xor(mx, o, 16));
    float sum = 0.f;
    for (int c = lid; c < 1024; c += 16) {
      float e = __expf(sc[r * 1024 + c] - mx);
      sc[r * 1024 + c] = e;
      sum += e;
    }
#pragma unroll
    for (int o = 8; o; o >>= 1) sum += __shfl_xor(sum, o, 16);
    float inv = 1.f / sum;
    for (int c = lid; c < 1024; c += 16)
      pb[r * LP + c] = (bf16_t)(sc[r * 1024 + c] * inv);
  }
  __syncthreads();

  // phase 3: O = P @ V ; 4 waves, one 16-wide d-tile each, K=1024
  if (wave < 4) {
    int d0 = wave * 16;
    const bf16_t* vb = Vt + ((size_t)bh * 64 + d0) * S;
    v8f a = {};
    for (int kc = 0; kc < 1024; kc += 32)
      a = wmma_bf16(load_frag(pb + kc, LP), load_frag(vb + kc, S), a);
#pragma unroll
    for (int r = 0; r < 8; ++r) {
      int m = r + half * 8;
      Out[((size_t)b * 1024 + q0 + m) * 768 + h * 64 + d0 + nl] = (bf16_t)a[r];
    }
  }
}

// ---------------------------------------------------------------------------
// Host orchestration
// ---------------------------------------------------------------------------
extern "C" void kernel_launch(void* const* d_in, const int* in_sizes, int n_in,
                              void* d_out, int out_size, void* d_ws, size_t ws_size,
                              hipStream_t stream) {
  (void)in_sizes; (void)n_in; (void)out_size; (void)ws_size;
  const int S = 1024, D = 768, H = 12, F = 3072, L = 12, V = 50257;
  const int M = 4 * S;  // 4096 token rows

  const int*   ids   = (const int*)  d_in[0];
  const float* tok   = (const float*)d_in[1];
  const float* pos   = (const float*)d_in[2];
  const float* ln1_g = (const float*)d_in[3];
  const float* ln1_b = (const float*)d_in[4];
  const float* qkv_w = (const float*)d_in[5];
  const float* qkv_b = (const float*)d_in[6];
  const float* out_w = (const float*)d_in[7];
  const float* out_b = (const float*)d_in[8];
  const float* ln2_g = (const float*)d_in[9];
  const float* ln2_b = (const float*)d_in[10];
  const float* fc1_w = (const float*)d_in[11];
  const float* fc1_b = (const float*)d_in[12];
  const float* fc2_w = (const float*)d_in[13];
  const float* fc2_b = (const float*)d_in[14];
  const float* lnf_g = (const float*)d_in[15];
  const float* lnf_b = (const float*)d_in[16];

  char* w = (char*)d_ws;
  auto take = [&](size_t bytes) -> char* {
    char* p = w;
    w += (bytes + 255) & ~(size_t)255;
    return p;
  };
  const size_t NQKV = (size_t)L * D * 3 * D;
  const size_t NOUT = (size_t)L * D * D;
  const size_t NFC1 = (size_t)L * D * F;
  const size_t NFC2 = (size_t)L * F * D;
  const size_t NTOK = (size_t)V * D;

  bf16_t* wqkvT = (bf16_t*)take(NQKV * 2);  // per layer [3D, D]
  bf16_t* woutT = (bf16_t*)take(NOUT * 2);  // per layer [D, D]
  bf16_t* wfc1T = (bf16_t*)take(NFC1 * 2);  // per layer [F, D]
  bf16_t* wfc2T = (bf16_t*)take(NFC2 * 2);  // per layer [D, F]
  bf16_t* wtok  = (bf16_t*)take(NTOK * 2);  // [V, D] (already N-major)
  float*  x     = (float*) take((size_t)M * D * 4);
  bf16_t* hbuf  = (bf16_t*)take((size_t)M * D * 2);
  bf16_t* qkvb  = (bf16_t*)take((size_t)M * 3 * D * 2);
  bf16_t* Qh    = (bf16_t*)take((size_t)M * D * 2);
  bf16_t* Kh    = (bf16_t*)take((size_t)M * D * 2);
  bf16_t* Vt    = (bf16_t*)take((size_t)M * D * 2);
  bf16_t* attb  = (bf16_t*)take((size_t)M * D * 2);
  bf16_t* ff1b  = (bf16_t*)take((size_t)M * F * 2);
  bf16_t* lnfb  = (bf16_t*)take((size_t)M * D * 2);

  const dim3 blk(256);
  // one-time (per launch) weight convert+transpose to [N,K] bf16
  for (int l = 0; l < L; ++l) {
    f32_to_bf16_t<<<dim3(3 * D / 32, D / 32), blk, 0, stream>>>(
        qkv_w + (size_t)l * D * 3 * D, wqkvT + (size_t)l * 3 * D * D, D, 3 * D);
    f32_to_bf16_t<<<dim3(D / 32, D / 32), blk, 0, stream>>>(
        out_w + (size_t)l * D * D, woutT + (size_t)l * D * D, D, D);
    f32_to_bf16_t<<<dim3(F / 32, D / 32), blk, 0, stream>>>(
        fc1_w + (size_t)l * D * F, wfc1T + (size_t)l * F * D, D, F);
    f32_to_bf16_t<<<dim3(D / 32, F / 32), blk, 0, stream>>>(
        fc2_w + (size_t)l * F * D, wfc2T + (size_t)l * D * F, F, D);
  }
  f32_to_bf16<<<dim3((unsigned)((NTOK + 255) / 256)), blk, 0, stream>>>(tok, wtok, NTOK);

  embed_kernel<<<M, blk, 0, stream>>>(ids, tok, pos, x);

  for (int l = 0; l < L; ++l) {
    layernorm_bf16<<<M, blk, 0, stream>>>(x, ln1_g + l * D, ln1_b + l * D, hbuf, D);
    gemm_bf16<1><<<dim3(3 * D / BN, M / BM), blk, 0, stream>>>(
        hbuf, wqkvT + (size_t)l * 3 * D * D, qkv_b + (size_t)l * 3 * D,
        nullptr, qkvb, M, 3 * D, D);
    reshape_qkv<<<M, blk, 0, stream>>>(qkvb, Qh, Kh, Vt);
    attn_kernel<<<dim3(S / 16, 4 * H), blk, 0, stream>>>(Qh, Kh, Vt, attb);
    gemm_bf16<2><<<dim3(D / BN, M / BM), blk, 0, stream>>>(
        attb, woutT + (size_t)l * D * D, out_b + (size_t)l * D, x, x, M, D, D);
    layernorm_bf16<<<M, blk, 0, stream>>>(x, ln2_g + l * D, ln2_b + l * D, hbuf, D);
    gemm_bf16<3><<<dim3(F / BN, M / BM), blk, 0, stream>>>(
        hbuf, wfc1T + (size_t)l * F * D, fc1_b + (size_t)l * F,
        nullptr, ff1b, M, F, D);
    gemm_bf16<2><<<dim3(D / BN, M / BM), blk, 0, stream>>>(
        ff1b, wfc2T + (size_t)l * D * F, fc2_b + (size_t)l * D, x, x, M, D, F);
  }

  layernorm_bf16<<<M, blk, 0, stream>>>(x, lnf_g, lnf_b, lnfb, D);
  // tied LM head: logits = lnf @ tok_embed^T
  gemm_bf16<0><<<dim3((V + BN - 1) / BN, M / BM), blk, 0, stream>>>(
      lnfb, wtok, nullptr, nullptr, d_out, M, V, D);
}
